// EGNN_5042291605587
// MI455X (gfx1250) — matrix operand order repaired and checked by reading, not compiled, not run on hardware
//
#include <hip/hip_runtime.h>
#include <hip/hip_bf16.h>

#define NODE_DIM 128
#define DEPTH 4
#define N_NODES 10000
#define N_EDGES 320000
#define WPB 8                       // waves per block
#define TPB (WPB * 32)

typedef __attribute__((ext_vector_type(16))) __bf16 v16bf;
typedef __attribute__((ext_vector_type(8)))  __bf16 v8bf;
typedef __attribute__((ext_vector_type(8)))  float  v8f;
typedef __attribute__((ext_vector_type(4)))  float  v4f;
typedef __attribute__((ext_vector_type(4)))  int    v4i;

__device__ __forceinline__ float silu(float t) {
  return t / (1.0f + __expf(-t));
}

__device__ __forceinline__ v8f wmma_bf16(v16bf a, v16bf b, v8f c) {
  // D = A(16x32) * B(32x16) + C, f32 accumulate
  return __builtin_amdgcn_wmma_f32_16x16x32_bf16(
      /*neg_a=*/false, a, /*neg_b=*/false, b,
      /*c_mod=*/(short)0, c, /*reuse_a=*/false, /*reuse_b=*/false);
}

// Build this lane's A fragment (16x32 bf16 layout) from a global f32 row.
__device__ __forceinline__ v16bf make_A_global(const float* __restrict__ row, int k0) {
  v4f a0 = *(const v4f*)(row + k0);
  v4f a1 = *(const v4f*)(row + k0 + 4);
  v4f b0 = *(const v4f*)(row + k0 + 16);
  v4f b1 = *(const v4f*)(row + k0 + 20);
  v16bf r;
#pragma unroll
  for (int j = 0; j < 4; ++j) {
    r[j]      = (__bf16)a0[j];
    r[4 + j]  = (__bf16)a1[j];
    r[8 + j]  = (__bf16)b0[j];
    r[12 + j] = (__bf16)b1[j];
  }
  return r;
}

// Same, but from a row-major bf16 row staged in LDS.
__device__ __forceinline__ v16bf make_A_lds(const __bf16* __restrict__ row, int k0) {
  v8bf a = *(const v8bf*)(row + k0);
  v8bf b = *(const v8bf*)(row + k0 + 16);
  v16bf r;
#pragma unroll
  for (int j = 0; j < 8; ++j) { r[j] = a[j]; r[8 + j] = b[j]; }
  return r;
}

// B fragment for (ks, nt) from the pre-swizzled LDS weight image: 32B per lane.
__device__ __forceinline__ v16bf load_B(const __bf16* __restrict__ sw, int ks, int nt, int lane) {
  return *(const v16bf*)(sw + ((((ks << 3) + nt) << 5) + lane) * 16);
}

// ---------------------------------------------------------------------------
// One-shot: swizzle a 128x128 f32 weight block (row-major [k][n]) into the
// WMMA B-operand lane layout as bf16, stored in global workspace.
//   dst[((ks*8+nt)*32+lane)*16+j] = W[(ks*32+(j>=8?16:0)+(lane>=16?8:0)+(j&7))*128 + nt*16+(lane&15)]
// ---------------------------------------------------------------------------
__global__ void swizzle_w(const float* __restrict__ W, __bf16* __restrict__ dst) {
  int idx = blockIdx.x * blockDim.x + threadIdx.x;
  if (idx < 16384) {
    int j    = idx & 15;
    int lane = (idx >> 4) & 31;
    int nt   = (idx >> 9) & 7;
    int ks   = idx >> 12;
    int k = ks * 32 + ((j >> 3) << 4) + ((lane >> 4) << 3) + (j & 7);
    int n = (nt << 4) + (lane & 15);
    dst[idx] = (__bf16)W[k * 128 + n];
  }
}

// Copy one pre-swizzled 32KB weight image global -> LDS via the CDNA5 async
// global->LDS data path (ASYNCcnt), falling back to a sync vector copy.
#define ASYNC_STAGE (__has_builtin(__builtin_amdgcn_global_load_async_to_lds_b128))

__device__ __forceinline__ void stage_copy(const __bf16* __restrict__ g,
                                           __bf16* __restrict__ l, int tid) {
#if ASYNC_STAGE
  typedef __attribute__((address_space(1))) v4i gv4i;
  typedef __attribute__((address_space(3))) v4i lv4i;
  for (int i = tid; i < 2048; i += TPB) {
    __builtin_amdgcn_global_load_async_to_lds_b128(
        (gv4i*)(g + (size_t)i * 8),
        (lv4i*)(l + (size_t)i * 8),
        0, 0);
  }
#else
  for (int i = tid; i < 2048; i += TPB)
    *(v8bf*)(l + (size_t)i * 8) = *(const v8bf*)(g + (size_t)i * 8);
#endif
}

__device__ __forceinline__ void stage_fence() {
#if ASYNC_STAGE
  asm volatile("s_wait_asynccnt 0x0" ::: "memory");
#endif
  __syncthreads();
}

// ---------------------------------------------------------------------------
// Edge kernel: one 16-edge M-tile per wave, WPB waves per block.
//   m = SiLU(SiLU([h_r, h_c, radial] W1 + b1) W2 + b2)
//   magg[row] += m (atomics); last layer also does the coord path.
// Weight images arrive pre-swizzled bf16 from workspace.
// ---------------------------------------------------------------------------
template<int LAST>
__global__ __launch_bounds__(TPB) void egnn_edge(
    const float* __restrict__ h, const float* __restrict__ x,
    const int* __restrict__ edges,
    const __bf16* __restrict__ gW1A, const __bf16* __restrict__ gW1B,
    const __bf16* __restrict__ gW2,  const __bf16* __restrict__ gCW1,
    const float* __restrict__ w1rad, const float* __restrict__ b1,
    const float* __restrict__ b2,
    const float* __restrict__ cb1,   const float* __restrict__ cw2,
    float* __restrict__ magg, float* __restrict__ aggx, float* __restrict__ cnt)
{
  __shared__ __align__(32) __bf16 sW1A[16384];
  __shared__ __align__(32) __bf16 sW1B[16384];
  __shared__ __align__(32) __bf16 sW2 [16384];
  __shared__ __align__(32) __bf16 sCW1[LAST ? 16384 : 32];
  __shared__ __align__(32) __bf16 scr[WPB][16 * 128];      // per-wave transpose tile
  __shared__ float sW1r[128], sB1[128], sB2[128];
  __shared__ float sCB1[LAST ? 128 : 4], sCW2[LAST ? 128 : 4];

  const int tid = threadIdx.x;
  stage_copy(gW1A, sW1A, tid);
  stage_copy(gW1B, sW1B, tid);
  stage_copy(gW2,  sW2,  tid);
  if (LAST) stage_copy(gCW1, sCW1, tid);
  for (int i = tid; i < 128; i += TPB) {
    sW1r[i] = w1rad[i];
    sB1[i] = b1[i];
    sB2[i] = b2[i];
    if (LAST) { sCB1[i] = cb1[i]; sCW2[i] = cw2[i]; }
  }
  stage_fence();

  const int wv   = tid >> 5;
  const int lane = tid & 31;
  const int tile = blockIdx.x * WPB + wv;
  const int e0   = tile * 16;
  if (e0 >= N_EDGES) return;
  const int hi = lane >> 4;
  const int lm = lane & 15;

  const int myRow = edges[e0 + lm];
  const int myCol = edges[N_EDGES + e0 + lm];
  const float* aR = h + (size_t)myRow * NODE_DIM;
  const float* aC = h + (size_t)myCol * NODE_DIM;

  v8f acc[8] = {};

  // ---- GEMM1: h_row * W1[0:128]  +  h_col * W1[128:256] ----
#pragma unroll
  for (int ks = 0; ks < 4; ++ks) {
    v16bf a = make_A_global(aR, ks * 32 + hi * 8);
#pragma unroll
    for (int nt = 0; nt < 8; ++nt)
      acc[nt] = wmma_bf16(a, load_B(sW1A, ks, nt, lane), acc[nt]);
  }
#pragma unroll
  for (int ks = 0; ks < 4; ++ks) {
    v16bf a = make_A_global(aC, ks * 32 + hi * 8);
#pragma unroll
    for (int nt = 0; nt < 8; ++nt)
      acc[nt] = wmma_bf16(a, load_B(sW1B, ks, nt, lane), acc[nt]);
  }

  // ---- rank-1 radial term + bias + SiLU; write bf16 tile for GEMM2 ----
  float rad[8];
#pragma unroll
  for (int v = 0; v < 8; ++v) {
    int m = v + hi * 8;
    int r = edges[e0 + m], c = edges[N_EDGES + e0 + m];
    float dx = x[r * 3 + 0] - x[c * 3 + 0];
    float dy = x[r * 3 + 1] - x[c * 3 + 1];
    float dz = x[r * 3 + 2] - x[c * 3 + 2];
    rad[v] = dx * dx + dy * dy + dz * dz;
  }
  __bf16* S = scr[wv];
#pragma unroll
  for (int nt = 0; nt < 8; ++nt) {
    float w256 = sW1r[nt * 16 + lm];
    float bb   = sB1 [nt * 16 + lm];
#pragma unroll
    for (int v = 0; v < 8; ++v) {
      float t = silu(acc[nt][v] + rad[v] * w256 + bb);
      S[(v + hi * 8) * 128 + nt * 16 + lm] = (__bf16)t;   // C-layout -> row-major
    }
  }

  // ---- GEMM2: m1 * W2 ----
  v8f a2[8] = {};
  const __bf16* srow = S + lm * 128;
#pragma unroll
  for (int ks = 0; ks < 4; ++ks) {
    v16bf a = make_A_lds(srow, ks * 32 + hi * 8);
#pragma unroll
    for (int nt = 0; nt < 8; ++nt)
      a2[nt] = wmma_bf16(a, load_B(sW2, ks, nt, lane), a2[nt]);
  }
#pragma unroll
  for (int nt = 0; nt < 8; ++nt)
#pragma unroll
    for (int v = 0; v < 8; ++v)
      a2[nt][v] = silu(a2[nt][v] + sB2[nt * 16 + lm]);

  // ---- scatter m into magg[row] (L2-resident atomics) ----
  int rows8[8];
#pragma unroll
  for (int v = 0; v < 8; ++v) rows8[v] = edges[e0 + v + hi * 8];
#pragma unroll
  for (int nt = 0; nt < 8; ++nt) {
    int n = nt * 16 + lm;
#pragma unroll
    for (int v = 0; v < 8; ++v)
      atomicAdd(&magg[(size_t)rows8[v] * NODE_DIM + n], a2[nt][v]);
  }

  if (LAST) {
    // ---- coord path: w = SiLU(m CW1 + cb1) . cw2 ----
#pragma unroll
    for (int nt = 0; nt < 8; ++nt)
#pragma unroll
      for (int v = 0; v < 8; ++v)
        S[(v + hi * 8) * 128 + nt * 16 + lm] = (__bf16)a2[nt][v];

    v8f a3[8] = {};
#pragma unroll
    for (int ks = 0; ks < 4; ++ks) {
      v16bf a = make_A_lds(srow, ks * 32 + hi * 8);
#pragma unroll
      for (int nt = 0; nt < 8; ++nt)
        a3[nt] = wmma_bf16(a, load_B(sCW1, ks, nt, lane), a3[nt]);
    }
    float s[8] = {0.f, 0.f, 0.f, 0.f, 0.f, 0.f, 0.f, 0.f};
#pragma unroll
    for (int nt = 0; nt < 8; ++nt) {
      int n = nt * 16 + lm;
      float c2 = sCW2[n], cb = sCB1[n];
#pragma unroll
      for (int v = 0; v < 8; ++v)
        s[v] += silu(a3[nt][v] + cb) * c2;
    }
    // reduce across the 16 lanes of each half (xor masks < 16 stay in-half)
#pragma unroll
    for (int off = 8; off >= 1; off >>= 1)
#pragma unroll
      for (int v = 0; v < 8; ++v)
        s[v] += __shfl_xor(s[v], off, 32);

    if (lm == 0) {
#pragma unroll
      for (int v = 0; v < 8; ++v) {
        int m = v + hi * 8;
        int r = edges[e0 + m], c = edges[N_EDGES + e0 + m];
        float dx = x[r * 3 + 0] - x[c * 3 + 0];
        float dy = x[r * 3 + 1] - x[c * 3 + 1];
        float dz = x[r * 3 + 2] - x[c * 3 + 2];
        float w = s[v];
        atomicAdd(&aggx[r * 3 + 0], dx * w);
        atomicAdd(&aggx[r * 3 + 1], dy * w);
        atomicAdd(&aggx[r * 3 + 2], dz * w);
        atomicAdd(&cnt[r], 1.0f);
      }
    }
  }
}

// ---------------------------------------------------------------------------
// Node kernel: h += SiLU([h, magg] W1 + b1) W2 + b2  (in-place on h)
// ---------------------------------------------------------------------------
__global__ __launch_bounds__(TPB) void egnn_node(
    float* __restrict__ h, const float* __restrict__ magg,
    const __bf16* __restrict__ gW1A, const __bf16* __restrict__ gW1B,
    const __bf16* __restrict__ gW2,
    const float* __restrict__ b1, const float* __restrict__ b2)
{
  __shared__ __align__(32) __bf16 sW1A[16384];
  __shared__ __align__(32) __bf16 sW1B[16384];
  __shared__ __align__(32) __bf16 sW2 [16384];
  __shared__ __align__(32) __bf16 scr[WPB][16 * 128];
  __shared__ float sB1[128], sB2[128];

  const int tid = threadIdx.x;
  stage_copy(gW1A, sW1A, tid);
  stage_copy(gW1B, sW1B, tid);
  stage_copy(gW2,  sW2,  tid);
  for (int i = tid; i < 128; i += TPB) { sB1[i] = b1[i]; sB2[i] = b2[i]; }
  stage_fence();

  const int wv   = tid >> 5;
  const int lane = tid & 31;
  const int tile = blockIdx.x * WPB + wv;
  const int n0   = tile * 16;
  if (n0 >= N_NODES) return;
  const int hi = lane >> 4;
  const int lm = lane & 15;

  const float* aH = h    + (size_t)(n0 + lm) * NODE_DIM;
  const float* aM = magg + (size_t)(n0 + lm) * NODE_DIM;

  v8f acc[8] = {};
#pragma unroll
  for (int ks = 0; ks < 4; ++ks) {
    v16bf a = make_A_global(aH, ks * 32 + hi * 8);
#pragma unroll
    for (int nt = 0; nt < 8; ++nt)
      acc[nt] = wmma_bf16(a, load_B(sW1A, ks, nt, lane), acc[nt]);
  }
#pragma unroll
  for (int ks = 0; ks < 4; ++ks) {
    v16bf a = make_A_global(aM, ks * 32 + hi * 8);
#pragma unroll
    for (int nt = 0; nt < 8; ++nt)
      acc[nt] = wmma_bf16(a, load_B(sW1B, ks, nt, lane), acc[nt]);
  }

  __bf16* S = scr[wv];
#pragma unroll
  for (int nt = 0; nt < 8; ++nt)
#pragma unroll
    for (int v = 0; v < 8; ++v)
      S[(v + hi * 8) * 128 + nt * 16 + lm] =
          (__bf16)silu(acc[nt][v] + sB1[nt * 16 + lm]);

  v8f a2[8] = {};
  const __bf16* srow = S + lm * 128;
#pragma unroll
  for (int ks = 0; ks < 4; ++ks) {
    v16bf a = make_A_lds(srow, ks * 32 + hi * 8);
#pragma unroll
    for (int nt = 0; nt < 8; ++nt)
      a2[nt] = wmma_bf16(a, load_B(sW2, ks, nt, lane), a2[nt]);
  }

#pragma unroll
  for (int nt = 0; nt < 8; ++nt) {
    int n = nt * 16 + lm;
#pragma unroll
    for (int v = 0; v < 8; ++v) {
      size_t off = (size_t)(n0 + v + hi * 8) * NODE_DIM + n;
      h[off] = h[off] + a2[nt][v] + sB2[n];   // residual
    }
  }
}

// ---------------------------------------------------------------------------
__global__ void zero_f32(float* __restrict__ p, int n) {
  int i = blockIdx.x * blockDim.x + threadIdx.x;
  if (i < n) p[i] = 0.0f;
}

__global__ void coord_fin(float* __restrict__ xOut, const float* __restrict__ x0,
                          const float* __restrict__ aggx, const float* __restrict__ cnt) {
  int i = blockIdx.x * blockDim.x + threadIdx.x;
  if (i < N_NODES * 3) {
    int node = i / 3;
    float c = fmaxf(cnt[node], 1.0f);
    xOut[i] = x0[i] + aggx[i] / c;
  }
}

// ---------------------------------------------------------------------------
extern "C" void kernel_launch(void* const* d_in, const int* in_sizes, int n_in,
                              void* d_out, int out_size, void* d_ws, size_t ws_size,
                              hipStream_t stream) {
  (void)in_sizes; (void)n_in; (void)out_size; (void)ws_size;
  const float* h0  = (const float*)d_in[0];
  const float* x0  = (const float*)d_in[1];
  const int*   ed  = (const int*)d_in[2];
  const float* ew1 = (const float*)d_in[3];
  const float* eb1 = (const float*)d_in[4];
  const float* ew2 = (const float*)d_in[5];
  const float* eb2 = (const float*)d_in[6];
  const float* nw1 = (const float*)d_in[7];
  const float* nb1 = (const float*)d_in[8];
  const float* nw2 = (const float*)d_in[9];
  const float* nb2 = (const float*)d_in[10];
  const float* cw1 = (const float*)d_in[11];
  const float* cb1 = (const float*)d_in[12];
  const float* cw2 = (const float*)d_in[13];

  float* hBuf = (float*)d_out;                       // [N, 128]
  float* xBuf = hBuf + (size_t)N_NODES * NODE_DIM;   // [N, 3]
  float* magg = (float*)d_ws;                        // [N, 128]
  float* aggx = magg + (size_t)N_NODES * NODE_DIM;   // [N, 3]
  float* cnt  = aggx + (size_t)N_NODES * 3;          // [N]
  // bf16 swizzled weight images, 16384 elems (32 KB) per slot:
  //   slot layer*6 + {0:eW1A, 1:eW1B, 2:eW2, 3:nW1A, 4:nW1B, 5:nW2}, slot 24: CW1
  __bf16* swz = (__bf16*)(cnt + N_NODES);

  (void)hipMemcpyAsync(hBuf, h0, sizeof(float) * (size_t)N_NODES * NODE_DIM,
                       hipMemcpyDeviceToDevice, stream);

  // ---- one-shot weight swizzle+convert into workspace ----
  for (int l = 0; l < DEPTH; ++l) {
    const float* W1 = ew1 + (size_t)l * 257 * 128;
    const float* W2 = ew2 + (size_t)l * 128 * 128;
    const float* V1 = nw1 + (size_t)l * 256 * 128;
    const float* V2 = nw2 + (size_t)l * 128 * 128;
    swizzle_w<<<64, 256, 0, stream>>>(W1,             swz + (size_t)(l * 6 + 0) * 16384);
    swizzle_w<<<64, 256, 0, stream>>>(W1 + 128 * 128, swz + (size_t)(l * 6 + 1) * 16384);
    swizzle_w<<<64, 256, 0, stream>>>(W2,             swz + (size_t)(l * 6 + 2) * 16384);
    swizzle_w<<<64, 256, 0, stream>>>(V1,             swz + (size_t)(l * 6 + 3) * 16384);
    swizzle_w<<<64, 256, 0, stream>>>(V1 + 128 * 128, swz + (size_t)(l * 6 + 4) * 16384);
    swizzle_w<<<64, 256, 0, stream>>>(V2,             swz + (size_t)(l * 6 + 5) * 16384);
  }
  swizzle_w<<<64, 256, 0, stream>>>(cw1, swz + (size_t)24 * 16384);

  const int ZN = N_NODES * (NODE_DIM + 3 + 1);        // magg + aggx + cnt
  const int edgeBlocks = (N_EDGES / 16 + WPB - 1) / WPB;   // 2500
  const int nodeBlocks = (N_NODES / 16 + WPB - 1) / WPB;   // 79

  for (int layer = 0; layer < DEPTH; ++layer) {
    zero_f32<<<(ZN + 255) / 256, 256, 0, stream>>>(magg, ZN);

    const __bf16* eA = swz + (size_t)(layer * 6 + 0) * 16384;
    const __bf16* eB = swz + (size_t)(layer * 6 + 1) * 16384;
    const __bf16* eC = swz + (size_t)(layer * 6 + 2) * 16384;
    const __bf16* nA = swz + (size_t)(layer * 6 + 3) * 16384;
    const __bf16* nB = swz + (size_t)(layer * 6 + 4) * 16384;
    const __bf16* nC = swz + (size_t)(layer * 6 + 5) * 16384;
    const __bf16* cA = swz + (size_t)24 * 16384;
    const float* W1rad = ew1 + (size_t)layer * 257 * 128 + 256 * 128;

    if (layer == DEPTH - 1) {
      egnn_edge<1><<<edgeBlocks, TPB, 0, stream>>>(
          hBuf, x0, ed, eA, eB, eC, cA, W1rad,
          eb1 + layer * 128, eb2 + layer * 128, cb1, cw2, magg, aggx, cnt);
      coord_fin<<<(N_NODES * 3 + 255) / 256, 256, 0, stream>>>(xBuf, x0, aggx, cnt);
    } else {
      egnn_edge<0><<<edgeBlocks, TPB, 0, stream>>>(
          hBuf, x0, ed, eA, eB, eC, cA, W1rad,
          eb1 + layer * 128, eb2 + layer * 128, cb1, cw2, magg, aggx, cnt);
    }

    egnn_node<<<nodeBlocks, TPB, 0, stream>>>(
        hBuf, magg, nA, nB, nC, nb1 + layer * 128, nb2 + layer * 128);
  }
}
